// TransformerEncoder_50337016709839
// MI455X (gfx1250) — compile-verified, matrix operand order checked by reading, not run on hardware
//
#include <hip/hip_runtime.h>

// ---------------------------------------------------------------------------
// Model constants (from the reference)
// ---------------------------------------------------------------------------
constexpr int Bb   = 2;
constexpr int Nseq = 2048;
constexpr int Dm   = 512;
constexpr int Hh   = 8;
constexpr int HD   = 64;
constexpr int Ff   = 2048;
constexpr int Mrows = Bb * Nseq;     // 4096
constexpr float QK_SCALE = 0.125f;   // HD^-0.5
constexpr float LN_EPS   = 1e-5f;

typedef __attribute__((ext_vector_type(16))) _Float16 v16h;
typedef __attribute__((ext_vector_type(8)))  _Float16 v8h;
typedef __attribute__((ext_vector_type(8)))  float    v8f;

static __device__ __forceinline__ v8f wmma_f16(v16h a, v16h b, v8f c) {
  return __builtin_amdgcn_wmma_f32_16x16x32_f16(false, a, false, b, (short)0, c,
                                                false, false);
}

static __device__ __forceinline__ v16h cat8(v8h lo, v8h hi) {
  return __builtin_shufflevector(lo, hi, 0,1,2,3,4,5,6,7,8,9,10,11,12,13,14,15);
}

// ---- DPP row-rotate reductions (16-lane rows; stays within wave32 halves) ----
template <int N>
static __device__ __forceinline__ float dpp_ror_f(float x) {
  return __int_as_float(
      __builtin_amdgcn_mov_dpp(__float_as_int(x), 0x120 | N, 0xF, 0xF, true));
}
static __device__ __forceinline__ float rowmax16(float v) {
  v = fmaxf(v, dpp_ror_f<1>(v));
  v = fmaxf(v, dpp_ror_f<2>(v));
  v = fmaxf(v, dpp_ror_f<4>(v));
  v = fmaxf(v, dpp_ror_f<8>(v));
  return v;
}
static __device__ __forceinline__ float rowsum16(float v) {
  v += dpp_ror_f<1>(v);
  v += dpp_ror_f<2>(v);
  v += dpp_ror_f<4>(v);
  v += dpp_ror_f<8>(v);
  return v;
}

// ---- CDNA5 async global->LDS copy (ASYNCcnt-tracked, bypasses VGPRs) ----
static __device__ __forceinline__ void async_copy_b128(const _Float16* g,
                                                       _Float16* l) {
  unsigned lds = (unsigned)(uintptr_t)(void*)l;
  unsigned long long ga = (unsigned long long)(uintptr_t)(const void*)g;
  asm volatile("global_load_async_to_lds_b128 %0, %1, off"
               :: "v"(lds), "v"(ga) : "memory");
}
static __device__ __forceinline__ void wait_async_le8() {
  asm volatile("s_wait_asynccnt 0x8" ::: "memory");
}
static __device__ __forceinline__ void wait_async_0() {
  asm volatile("s_wait_asynccnt 0x0" ::: "memory");
}

// ---------------------------------------------------------------------------
// f32 -> f16 conversion (weights once per launch; initial activations)
// n must be a multiple of 2048.
// ---------------------------------------------------------------------------
__global__ __launch_bounds__(256)
void cvt_kernel(const float* __restrict__ in, _Float16* __restrict__ out)
{
  const size_t i = ((size_t)blockIdx.x * 256 + threadIdx.x) * 8;
  const float4* p = reinterpret_cast<const float4*>(in + i);
  float4 a = p[0], b = p[1];
  v8h h = { (_Float16)a.x, (_Float16)a.y, (_Float16)a.z, (_Float16)a.w,
            (_Float16)b.x, (_Float16)b.y, (_Float16)b.z, (_Float16)b.w };
  *(v8h*)(out + i) = h;
}

// ---------------------------------------------------------------------------
// GEMM: Out = epilogue( A[M x K](f16) * W[Nn x K](f16)^T + bias )
//  OMODE 0: f32 row-major; 1: f16 [b,h,n,hd] (scaled); 2: f16 [b,h,hd,n];
//  OMODE 3: f16 row-major.
// 128x128 tile, BK=64, double-buffered global_load_async_to_lds_b128 staging.
// 256 threads = 8 waves; wave computes 32x64 -> 16 WMMAs per stage.
// ---------------------------------------------------------------------------
template <int OMODE, bool RELU, bool ADD_RES>
__global__ __launch_bounds__(256)
void gemm_kernel(const _Float16* __restrict__ A, const _Float16* __restrict__ W,
                 const float* __restrict__ bias, const float* __restrict__ Res,
                 void* __restrict__ Out, int M, int Nn, int K, float scale)
{
  constexpr int BM = 128, BN = 128, BK = 64, LDT = BK + 8;  // 72 halves (144B)
  constexpr int TSZ = BM * LDT;
  __shared__ alignas(16) _Float16 As[2 * TSZ];
  __shared__ alignas(16) _Float16 Bs[2 * TSZ];

  const int n0   = blockIdx.x * BN;
  const int m0   = blockIdx.y * BM;
  const int t    = threadIdx.x;
  const int lane = t & 31;
  const int wid  = t >> 5;
  const int lr   = lane & 15;
  const int hiq  = lane >> 4;
  const int hic  = hiq * 8;            // K sub-offset per ISA fragment layout
  const int wm   = (wid >> 1) * 32;
  const int wn   = (wid & 1) * 64;

  // Loader: 1024 16B chunks per 128x64 tile; each thread copies 4 per matrix.
  const int cid0 = t * 4;

  auto issue_tile = [&](int buf, int k0) {
    #pragma unroll
    for (int i = 0; i < 4; ++i) {
      const int cid = cid0 + i;
      const int row = cid >> 3;
      const int ch  = (cid & 7) * 8;
      async_copy_b128(A + (size_t)(m0 + row) * K + k0 + ch,
                      &As[buf * TSZ + row * LDT + ch]);
      async_copy_b128(W + (size_t)(n0 + row) * K + k0 + ch,
                      &Bs[buf * TSZ + row * LDT + ch]);
    }
  };

  issue_tile(0, 0);   // prologue: 8 async b128 per wave

  v8f acc[2][4];
  #pragma unroll
  for (int i = 0; i < 2; ++i)
    #pragma unroll
    for (int j = 0; j < 4; ++j) acc[i][j] = {};

  const int nsteps = K / BK;
  for (int it = 0; it < nsteps; ++it) {
    const int cur = it & 1, nxt = cur ^ 1;
    if ((it + 1) < nsteps) {
      issue_tile(nxt, (it + 1) * BK);
      wait_async_le8();   // async loads complete in order -> tile `cur` landed
    } else {
      wait_async_0();
    }
    __syncthreads();

    const _Float16* Ac = &As[cur * TSZ];
    const _Float16* Wc = &Bs[cur * TSZ];
    #pragma unroll
    for (int kc = 0; kc < 2; ++kc) {
      v16h af[2], bf[4];
      #pragma unroll
      for (int mt = 0; mt < 2; ++mt) {
        const _Float16* p = &Ac[(wm + mt * 16 + lr) * LDT + kc * 32 + hic];
        af[mt] = cat8(*(const v8h*)p, *(const v8h*)(p + 16));
      }
      #pragma unroll
      for (int nt = 0; nt < 4; ++nt) {
        const _Float16* p = &Wc[(wn + nt * 16 + lr) * LDT + kc * 32 + hic];
        bf[nt] = cat8(*(const v8h*)p, *(const v8h*)(p + 16));
      }
      #pragma unroll
      for (int mt = 0; mt < 2; ++mt)
        #pragma unroll
        for (int nt = 0; nt < 4; ++nt)
          acc[mt][nt] = wmma_f16(af[mt], bf[nt], acc[mt][nt]);
    }
    __syncthreads();
  }

  // ---- epilogue ----
  #pragma unroll
  for (int mt = 0; mt < 2; ++mt) {
    #pragma unroll
    for (int nt = 0; nt < 4; ++nt) {
      #pragma unroll
      for (int r = 0; r < 8; ++r) {
        const int m = m0 + wm + mt * 16 + r + (hiq << 3);
        const int n = n0 + wn + nt * 16 + lr;
        float v = acc[mt][nt][r] + bias[n];
        if (RELU)    v = fmaxf(v, 0.0f);
        if (ADD_RES) v += Res[(size_t)m * Nn + n];
        if (OMODE == 0) {
          ((float*)Out)[(size_t)m * Nn + n] = v;
        } else if (OMODE == 3) {
          ((_Float16*)Out)[(size_t)m * Nn + n] = (_Float16)v;
        } else {
          const int bi = m >> 11, ns = m & (Nseq - 1);
          const int hh = n >> 6,  d  = n & (HD - 1);
          const _Float16 hv = (_Float16)(v * scale);
          if (OMODE == 1)   // [b, h, n, hd]
            ((_Float16*)Out)[(((size_t)(bi * Hh + hh)) * Nseq + ns) * HD + d] = hv;
          else              // [b, h, hd, n]
            ((_Float16*)Out)[(((size_t)(bi * Hh + hh)) * HD + d) * Nseq + ns] = hv;
        }
      }
    }
  }
}

// ---------------------------------------------------------------------------
// Flash attention. Q,K: [b,h,n,hd] f16 (Q pre-scaled). V: [b,h,hd,n] f16.
// 128 threads = 4 waves; each wave owns 16 query rows. K/V tiles staged
// cooperatively via double-buffered async copies; softmax via DPP row_ror.
// Output f16 [B*N, D] (concat heads).
// ---------------------------------------------------------------------------
__global__ __launch_bounds__(128)
void flash_kernel(const _Float16* __restrict__ Q, const _Float16* __restrict__ Kh,
                  const _Float16* __restrict__ VT, _Float16* __restrict__ O)
{
  constexpr int LDT = 72;   // half stride for 64-wide tiles (144B rows)
  __shared__ alignas(16) _Float16 Ks[2][64 * LDT];
  __shared__ alignas(16) _Float16 Vs[2][64 * LDT];
  __shared__ alignas(16) _Float16 Plds[4][16 * LDT];

  const int bh  = blockIdx.y;
  const int bi  = bh >> 3, hh = bh & 7;
  const int wid = threadIdx.x >> 5, lane = threadIdx.x & 31;
  const int lr  = lane & 15, hiq = lane >> 4, c = hiq * 8;
  const int q0  = blockIdx.x * 64 + wid * 16;

  const _Float16* Qb = Q  + (size_t)bh * Nseq * HD;
  const _Float16* Kb = Kh + (size_t)bh * Nseq * HD;
  const _Float16* Vb = VT + (size_t)bh * HD * Nseq;

  // Loader: 512 16B chunks per 64x64 tile; each thread copies 4 per matrix.
  const int cid0 = threadIdx.x * 4;
  auto issue_tile = [&](int buf, int j0) {
    #pragma unroll
    for (int i = 0; i < 4; ++i) {
      const int cid = cid0 + i;
      const int row = cid >> 3;          // 0..63
      const int ch  = (cid & 7) * 8;     // 0..56 halves
      async_copy_b128(Kb + (size_t)(j0 + row) * HD + ch,
                      &Ks[buf][row * LDT + ch]);
      async_copy_b128(Vb + (size_t)row * Nseq + j0 + ch,
                      &Vs[buf][row * LDT + ch]);
    }
  };

  issue_tile(0, 0);

  // Q fragments: K-dim = HD = 64 -> two 16x32 chunks, resident all kernel.
  v16h qf[2];
  #pragma unroll
  for (int kc = 0; kc < 2; ++kc) {
    const _Float16* p = Qb + (size_t)(q0 + lr) * HD + kc * 32 + c;
    qf[kc] = cat8(*(const v8h*)p, *(const v8h*)(p + 16));
  }

  float rmax[8], rsum[8];
  #pragma unroll
  for (int r = 0; r < 8; ++r) { rmax[r] = -1e30f; rsum[r] = 0.0f; }
  v8f accO[4];
  #pragma unroll
  for (int nt = 0; nt < 4; ++nt) accO[nt] = {};

  _Float16* Pw = &Plds[wid][0];

  constexpr int NSTEP = Nseq / 64;   // 32
  for (int it = 0; it < NSTEP; ++it) {
    const int cur = it & 1, nxt = cur ^ 1;
    if ((it + 1) < NSTEP) {
      issue_tile(nxt, (it + 1) * 64);
      wait_async_le8();
    } else {
      wait_async_0();
    }
    __syncthreads();

    // ---- S = Q * K^T (16 x 64 per wave) ----
    v8f s[4];
    #pragma unroll
    for (int nt = 0; nt < 4; ++nt) {
      const _Float16* kp = &Ks[cur][(nt * 16 + lr) * LDT + c];
      v16h kf0 = cat8(*(const v8h*)kp,        *(const v8h*)(kp + 16));
      v16h kf1 = cat8(*(const v8h*)(kp + 32), *(const v8h*)(kp + 48));
      v8f z = {};
      s[nt] = wmma_f16(qf[0], kf0, z);
      s[nt] = wmma_f16(qf[1], kf1, s[nt]);
    }

    // ---- online softmax (DPP row reductions, no LDS) ----
    #pragma unroll
    for (int r = 0; r < 8; ++r) {
      float mx = fmaxf(fmaxf(s[0][r], s[1][r]), fmaxf(s[2][r], s[3][r]));
      mx = rowmax16(mx);
      const float mnew  = fmaxf(rmax[r], mx);
      const float alpha = __expf(rmax[r] - mnew);
      rmax[r] = mnew;
      float sum = 0.0f;
      #pragma unroll
      for (int nt = 0; nt < 4; ++nt) {
        const float p = __expf(s[nt][r] - mnew);
        s[nt][r] = p;
        sum += p;
      }
      sum = rowsum16(sum);
      rsum[r] = rsum[r] * alpha + sum;
      #pragma unroll
      for (int nt = 0; nt < 4; ++nt) accO[nt][r] *= alpha;
    }

    // ---- P -> wave-private LDS (re-layout C-regs to A-fragments) ----
    #pragma unroll
    for (int nt = 0; nt < 4; ++nt)
      #pragma unroll
      for (int r = 0; r < 8; ++r)
        Pw[(r + 8 * hiq) * LDT + nt * 16 + lr] = (_Float16)s[nt][r];
    asm volatile("s_wait_dscnt 0" ::: "memory");

    // ---- O += P * V ----
    #pragma unroll
    for (int kc = 0; kc < 2; ++kc) {
      const _Float16* pp = Pw + lr * LDT + kc * 32 + c;
      v16h pf = cat8(*(const v8h*)pp, *(const v8h*)(pp + 16));
      #pragma unroll
      for (int nt = 0; nt < 4; ++nt) {
        const _Float16* vp = &Vs[cur][(nt * 16 + lr) * LDT + kc * 32 + c];
        v16h vf = cat8(*(const v8h*)vp, *(const v8h*)(vp + 16));
        accO[nt] = wmma_f16(pf, vf, accO[nt]);
      }
    }
    __syncthreads();
  }

  // ---- normalize + store f16 (concat heads) ----
  #pragma unroll
  for (int nt = 0; nt < 4; ++nt)
    #pragma unroll
    for (int r = 0; r < 8; ++r) {
      const int row = q0 + r + 8 * hiq;
      const int col = hh * HD + nt * 16 + lr;
      O[((size_t)(bi * Nseq + row)) * Dm + col] =
          (_Float16)(accO[nt][r] / rsum[r]);
    }
}

// ---------------------------------------------------------------------------
// LayerNorm over D=512; writes f32 (residual path) and f16 (WMMA path).
// ---------------------------------------------------------------------------
__global__ __launch_bounds__(256)
void layernorm_kernel(const float* __restrict__ X, const float* __restrict__ g,
                      const float* __restrict__ e, float* __restrict__ Out32,
                      _Float16* __restrict__ Out16)
{
  __shared__ float red[256];
  const int row = blockIdx.x;
  const int t   = threadIdx.x;
  const float* xr = X + (size_t)row * Dm;
  const float v0 = xr[t], v1 = xr[t + 256];

  red[t] = v0 + v1;
  __syncthreads();
  #pragma unroll
  for (int s = 128; s > 0; s >>= 1) {
    if (t < s) red[t] += red[t + s];
    __syncthreads();
  }
  const float mu = red[0] * (1.0f / Dm);
  __syncthreads();

  const float d0 = v0 - mu, d1 = v1 - mu;
  red[t] = d0 * d0 + d1 * d1;
  __syncthreads();
  #pragma unroll
  for (int s = 128; s > 0; s >>= 1) {
    if (t < s) red[t] += red[t + s];
    __syncthreads();
  }
  const float rs = rsqrtf(red[0] * (1.0f / Dm) + LN_EPS);

  const float o0 = d0 * rs * g[t]       + e[t];
  const float o1 = d1 * rs * g[t + 256] + e[t + 256];
  Out32[(size_t)row * Dm + t]       = o0;
  Out32[(size_t)row * Dm + t + 256] = o1;
  Out16[(size_t)row * Dm + t]       = (_Float16)o0;
  Out16[(size_t)row * Dm + t + 256] = (_Float16)o1;
}

// ---------------------------------------------------------------------------
// Host orchestration
// ---------------------------------------------------------------------------
extern "C" void kernel_launch(void* const* d_in, const int* in_sizes, int n_in,
                              void* d_out, int out_size, void* d_ws, size_t ws_size,
                              hipStream_t stream)
{
  const float* x  = (const float*)d_in[0];
  const float* Wq = (const float*)d_in[1];
  const float* bq = (const float*)d_in[2];
  const float* Wk = (const float*)d_in[3];
  const float* bk = (const float*)d_in[4];
  const float* Wv = (const float*)d_in[5];
  const float* bv = (const float*)d_in[6];
  const float* Wo = (const float*)d_in[7];
  const float* bo = (const float*)d_in[8];
  const float* W1 = (const float*)d_in[9];
  const float* b1 = (const float*)d_in[10];
  const float* W2 = (const float*)d_in[11];
  const float* b2 = (const float*)d_in[12];
  const float* g1 = (const float*)d_in[13];
  const float* e1 = (const float*)d_in[14];
  const float* g2 = (const float*)d_in[15];
  const float* e2 = (const float*)d_in[16];
  float* out = (float*)d_out;

  const size_t MB = 1048576;
  char* base = (char*)d_ws;
  float*    xw     = (float*)(base);                 //  8 MB  f32 [M x D]
  _Float16* x16    = (_Float16*)(base +  8 * MB);    //  4 MB
  float*    tmp    = (float*)(base + 12 * MB);       //  8 MB
  _Float16* attno  = (_Float16*)(base + 20 * MB);    //  4 MB
  _Float16* hbuf   = (_Float16*)(base + 24 * MB);    // 16 MB  [M x F]
  _Float16* q16    = (_Float16*)(base + 40 * MB);    //  4 MB
  _Float16* k16    = (_Float16*)(base + 44 * MB);    //  4 MB
  _Float16* v16    = (_Float16*)(base + 48 * MB);    //  4 MB
  _Float16* wq16   = (_Float16*)(base + 52 * MB);    //  2 MB  (L*D*D f16)
  _Float16* wk16   = (_Float16*)(base + 54 * MB);    //  2 MB
  _Float16* wv16   = (_Float16*)(base + 56 * MB);    //  2 MB
  _Float16* wo16   = (_Float16*)(base + 58 * MB);    //  2 MB
  _Float16* w116   = (_Float16*)(base + 60 * MB);    //  8 MB  (L*F*D f16)
  _Float16* w216   = (_Float16*)(base + 68 * MB);    //  8 MB  (L*D*F f16)

  const dim3 blk(256);
  const int LDD = 4 * Dm * Dm;      // 1,048,576
  const int LFD = 4 * Ff * Dm;      // 4,194,304

  // One-time f32 -> f16 conversions (weights + initial activations).
  cvt_kernel<<<dim3(LDD / 2048), blk, 0, stream>>>(Wq, wq16);
  cvt_kernel<<<dim3(LDD / 2048), blk, 0, stream>>>(Wk, wk16);
  cvt_kernel<<<dim3(LDD / 2048), blk, 0, stream>>>(Wv, wv16);
  cvt_kernel<<<dim3(LDD / 2048), blk, 0, stream>>>(Wo, wo16);
  cvt_kernel<<<dim3(LFD / 2048), blk, 0, stream>>>(W1, w116);
  cvt_kernel<<<dim3(LFD / 2048), blk, 0, stream>>>(W2, w216);
  cvt_kernel<<<dim3(Mrows * Dm / 2048), blk, 0, stream>>>(x, x16);
  hipMemcpyAsync(xw, x, (size_t)Mrows * Dm * sizeof(float),
                 hipMemcpyDeviceToDevice, stream);

  const dim3 gD(Dm / 128, Mrows / 128);   // (4, 32)
  const dim3 gF(Ff / 128, Mrows / 128);   // (16, 32)
  const dim3 gAttn(Nseq / 64, Bb * Hh);   // (32, 16)

  for (int l = 0; l < 4; ++l) {
    const _Float16* Wq_l = wq16 + (size_t)l * Dm * Dm;
    const _Float16* Wk_l = wk16 + (size_t)l * Dm * Dm;
    const _Float16* Wv_l = wv16 + (size_t)l * Dm * Dm;
    const _Float16* Wo_l = wo16 + (size_t)l * Dm * Dm;
    const _Float16* W1_l = w116 + (size_t)l * Ff * Dm;
    const _Float16* W2_l = w216 + (size_t)l * Dm * Ff;
    const float* bq_l = bq + (size_t)l * Dm;
    const float* bk_l = bk + (size_t)l * Dm;
    const float* bv_l = bv + (size_t)l * Dm;
    const float* bo_l = bo + (size_t)l * Dm;
    const float* b1_l = b1 + (size_t)l * Ff;
    const float* b2_l = b2 + (size_t)l * Dm;
    const float* g1_l = g1 + (size_t)l * Dm;
    const float* e1_l = e1 + (size_t)l * Dm;
    const float* g2_l = g2 + (size_t)l * Dm;
    const float* e2_l = e2 + (size_t)l * Dm;

    gemm_kernel<1, false, false><<<gD, blk, 0, stream>>>(
        x16, Wq_l, bq_l, nullptr, q16, Mrows, Dm, Dm, QK_SCALE);
    gemm_kernel<1, false, false><<<gD, blk, 0, stream>>>(
        x16, Wk_l, bk_l, nullptr, k16, Mrows, Dm, Dm, 1.0f);
    gemm_kernel<2, false, false><<<gD, blk, 0, stream>>>(
        x16, Wv_l, bv_l, nullptr, v16, Mrows, Dm, Dm, 1.0f);

    flash_kernel<<<gAttn, dim3(128), 0, stream>>>(q16, k16, v16, attno);

    gemm_kernel<0, false, true><<<gD, blk, 0, stream>>>(
        attno, Wo_l, bo_l, xw, tmp, Mrows, Dm, Dm, 1.0f);
    layernorm_kernel<<<dim3(Mrows), blk, 0, stream>>>(tmp, g1_l, e1_l, xw, x16);

    gemm_kernel<3, true, false><<<gF, blk, 0, stream>>>(
        x16, W1_l, b1_l, nullptr, hbuf, Mrows, Ff, Dm, 1.0f);
    gemm_kernel<0, false, true><<<gD, blk, 0, stream>>>(
        hbuf, W2_l, b2_l, xw, tmp, Mrows, Dm, Ff, 1.0f);
    layernorm_kernel<<<dim3(Mrows), blk, 0, stream>>>(tmp, g2_l, e2_l, xw, x16);
  }

  hipMemcpyAsync(out, xw, (size_t)Mrows * Dm * sizeof(float),
                 hipMemcpyDeviceToDevice, stream);
}
